// PropagateAttention_6914897347099
// MI455X (gfx1250) — matrix operand order repaired and checked by reading, not compiled, not run on hardware
//
#include <hip/hip_runtime.h>
#include <hip/hip_bf16.h>

// ---------------------------------------------------------------------------
// Problem constants
// ---------------------------------------------------------------------------
#define BATCH 2
#define NTOK  4096
#define CDIM  512
#define HEADS 8
#define DDIM  64
#define SCALE_QK 0.125f // 64^-0.5

typedef __attribute__((ext_vector_type(16))) _Float16 v16h;
typedef __attribute__((ext_vector_type(8)))  _Float16 v8h;
typedef __attribute__((ext_vector_type(8)))  float    v8f;

__device__ __forceinline__ v16h cat16(v8h lo, v8h hi) {
    v16h r;
#pragma unroll
    for (int i = 0; i < 8; ++i) { r[i] = lo[i]; r[i + 8] = hi[i]; }
    return r;
}

__device__ __forceinline__ v8f wmma16(v16h a, v16h b, v8f c) {
    return __builtin_amdgcn_wmma_f32_16x16x32_f16(
        /*neg_a=*/false, a, /*neg_b=*/false, b,
        /*c_mod=*/(short)0, c, /*reuse_a=*/false, /*reuse_b=*/false);
}

// Async global->LDS copy of 16 bytes (per lane). Tracked by ASYNCcnt.
__device__ __forceinline__ void async_copy_b128(unsigned lds_addr, uint64_t gaddr) {
    asm volatile("global_load_async_to_lds_b128 %0, %1, off"
                 :: "v"(lds_addr), "v"(gaddr) : "memory");
}
__device__ __forceinline__ void async_copy_b128_off16(unsigned lds_addr, uint64_t gaddr) {
    asm volatile("global_load_async_to_lds_b128 %0, %1, off offset:16"
                 :: "v"(lds_addr), "v"(gaddr) : "memory");
}
__device__ __forceinline__ void wait_async0() {
    asm volatile("s_wait_asynccnt 0x0" ::: "memory");
}

// Truncate a generic pointer to a shared-memory object to its LDS byte offset
// (flat LDS aperture: the offset is the low 32 bits of the address).
__device__ __forceinline__ unsigned lds_off(const void* p) {
    return (unsigned)(size_t)p;
}

// ---------------------------------------------------------------------------
// Kernel 1: input projection  Y = X @ W  (f32 in, f16 out in head layout)
//   X: [B*N, C] f32, W: [C, C] f32, Y: [B*H][N][D] f16
// Block: 256 threads (8 waves). Block tile 64(M) x 128(N), K-step 32.
// Each wave computes a 32x32 tile = 2x2 WMMA 16x16x32 tiles.
// ---------------------------------------------------------------------------
__global__ void __launch_bounds__(256)
proj_qkv_kernel(const float* __restrict__ X,
                const float* __restrict__ W,
                _Float16* __restrict__ Y)
{
    __shared__ __align__(16) _Float16 As[64][40];    // 64x32 + pad
    __shared__ __align__(16) _Float16 Bt[128][40];   // W tile transposed + pad

    const int t    = threadIdx.x;
    const int lane = t & 31;
    const int w    = t >> 5;
    const int wm   = w & 1;     // 2 M strips of 32
    const int wn   = w >> 1;    // 4 N strips of 32
    const int g    = lane >> 4; // half-wave select
    const int lh   = lane & 15;
    const int Mb   = blockIdx.x * 64;
    const int Nb   = blockIdx.y * 128;

    v8f acc[2][2] = {};

    for (int k0 = 0; k0 < CDIM; k0 += 32) {
        // stage A (64x32, f32 -> f16): 8 elems/thread
        {
            const int r = t >> 2, c = (t & 3) * 8;
            const float* src = X + (size_t)(Mb + r) * CDIM + k0 + c;
            if (k0 + 32 < CDIM) __builtin_prefetch(src + 32, 0, 0);
            v8h h;
#pragma unroll
            for (int i = 0; i < 8; ++i) h[i] = (_Float16)src[i];
            *(v8h*)&As[r][c] = h;
        }
        // stage Bt (32x128 -> transposed 128x32): 16 elems/thread
        {
            const int kk = t >> 3, n0 = (t & 7) * 16;
            const float* src = W + (size_t)(k0 + kk) * CDIM + Nb + n0;
            if (k0 + 32 < CDIM) __builtin_prefetch(src + 32 * CDIM, 0, 0);
#pragma unroll
            for (int i = 0; i < 16; ++i) Bt[n0 + i][kk] = (_Float16)src[i];
        }
        __syncthreads();

        v16h af[2], bf[2];
#pragma unroll
        for (int ms = 0; ms < 2; ++ms) {
            const int m = wm * 32 + ms * 16 + lh;
            af[ms] = cat16(*(const v8h*)&As[m][8 * g],
                           *(const v8h*)&As[m][8 * g + 16]);
        }
#pragma unroll
        for (int ns = 0; ns < 2; ++ns) {
            const int n = wn * 32 + ns * 16 + lh;
            bf[ns] = cat16(*(const v8h*)&Bt[n][8 * g],
                           *(const v8h*)&Bt[n][8 * g + 16]);
        }
#pragma unroll
        for (int ms = 0; ms < 2; ++ms)
#pragma unroll
            for (int ns = 0; ns < 2; ++ns)
                acc[ms][ns] = wmma16(af[ms], bf[ns], acc[ms][ns]);
        __syncthreads();
    }

    // epilogue: scatter into head layout [b*H + h][n][d] as f16
#pragma unroll
    for (int ms = 0; ms < 2; ++ms)
#pragma unroll
        for (int ns = 0; ns < 2; ++ns) {
            const int col = Nb + wn * 32 + ns * 16 + lh; // C/D layout: col = lane%16
            const int h   = col >> 6, d = col & 63;
#pragma unroll
            for (int r = 0; r < 8; ++r) {
                const int row = Mb + wm * 32 + ms * 16 + 8 * g + r;
                const int b   = row >> 12, n = row & (NTOK - 1);
                Y[(((size_t)(b * HEADS + h) * NTOK) + n) * DDIM + d] =
                    (_Float16)acc[ms][ns][r];
            }
        }
}

// ---------------------------------------------------------------------------
// Kernel 2: per-row scale (qc *= SCALE) or normalize (qs = qs/||qs|| * gamma[h])
// One wave per 64-element head row. grid.x = B*H*N/8, block 256 (8 waves).
// ---------------------------------------------------------------------------
__global__ void __launch_bounds__(256)
scale_norm_kernel(_Float16* __restrict__ q, const float* __restrict__ gamma,
                  float scale, int mode)
{
    const int w = threadIdx.x >> 5, lane = threadIdx.x & 31;
    const long row = (long)blockIdx.x * 8 + w;     // [0, B*H*N)
    _Float16* p = q + row * DDIM + lane * 2;
    float x0 = (float)p[0], x1 = (float)p[1];
    float s;
    if (mode == 0) {
        s = scale;
    } else {
        float ss = x0 * x0 + x1 * x1;
#pragma unroll
        for (int m = 16; m >= 1; m >>= 1) ss += __shfl_xor(ss, m, 32);
        const int h = (int)((row >> 12) & 7);      // row/N % H
        s = rsqrtf(ss) * gamma[h];
    }
    p[0] = (_Float16)(x0 * s);
    p[1] = (_Float16)(x1 * s);
}

// ---------------------------------------------------------------------------
// Kernel 3: flash attention over both streams.
// grid = (N/128, B*H). Block 256 = 8 waves; wave owns 16 query rows.
// Key tiles of 64. kc/ks staged row-major in LDS via async global->LDS B128
// (shared by all 8 waves); vc/vs staged transposed; P staged per-wave.
// Scales already folded into qc (SCALE) and qs (gamma/||qs||).
// ---------------------------------------------------------------------------
__global__ void __launch_bounds__(256)
attn_kernel(const _Float16* __restrict__ qc, const _Float16* __restrict__ qs,
            const _Float16* __restrict__ kc, const _Float16* __restrict__ ks,
            const _Float16* __restrict__ vc, const _Float16* __restrict__ vs,
            _Float16* __restrict__ oc, _Float16* __restrict__ osb)
{
    __shared__ __align__(16) _Float16 kcS[64][72];       // [key][d] row-major
    __shared__ __align__(16) _Float16 ksS[64][72];
    __shared__ __align__(16) _Float16 vcT[64][72];       // [d][key] transposed
    __shared__ __align__(16) _Float16 vsT[64][72];
    __shared__ __align__(16) _Float16 Pld[8][16][72];    // per-wave P strip

    const int t = threadIdx.x, lane = t & 31, w = t >> 5;
    const int g = lane >> 4, lh = lane & 15;
    const int bh = blockIdx.y;
    const int h = bh & 7, b = bh >> 3;
    const int q0 = blockIdx.x * 128 + w * 16;

    const size_t base = (size_t)bh * NTOK * DDIM;
    const _Float16* qcb = qc + base;
    const _Float16* qsb = qs + base;
    const _Float16* kcb = kc + base;
    const _Float16* ksb = ks + base;
    const _Float16* vcb = vc + base;
    const _Float16* vsb = vs + base;

    // Q fragments (A layout), hoisted across all key tiles
    v16h qcf[2], qsf[2];
    {
        const _Float16* rc = qcb + (size_t)(q0 + lh) * DDIM;
        const _Float16* rs = qsb + (size_t)(q0 + lh) * DDIM;
#pragma unroll
        for (int kk = 0; kk < 2; ++kk) {
            const int off = kk * 32 + 8 * g;
            qcf[kk] = cat16(*(const v8h*)(rc + off), *(const v8h*)(rc + off + 16));
            qsf[kk] = cat16(*(const v8h*)(rs + off), *(const v8h*)(rs + off + 16));
        }
    }

    float mrow[8], lrow[8];
#pragma unroll
    for (int r = 0; r < 8; ++r) { mrow[r] = -1e30f; lrow[r] = 0.0f; }
    v8f accc[4] = {};
    v8f accs[4] = {};

    for (int kt = 0; kt < NTOK; kt += 64) {
        __syncthreads(); // previous tile's LDS reads done
        // stage tiles: thread t handles key = t/4, 16 d-elems at (t%4)*16
        {
            const int key = t >> 2, d0 = (t & 3) * 16;
            // kc/ks: pure f16 row-major copies -> async global->LDS (32 B each)
            const uint64_t gkc = (uint64_t)(kcb + (size_t)(kt + key) * DDIM + d0);
            const uint64_t gks = (uint64_t)(ksb + (size_t)(kt + key) * DDIM + d0);
            const unsigned lkc = lds_off(&kcS[key][d0]);
            const unsigned lks = lds_off(&ksS[key][d0]);
            async_copy_b128(lkc, gkc);
            async_copy_b128_off16(lkc, gkc);
            async_copy_b128(lks, gks);
            async_copy_b128_off16(lks, gks);
            // vc/vs: transpose while staging (through VGPRs)
            const _Float16* s1 = vcb + (size_t)(kt + key) * DDIM + d0;
            const _Float16* s2 = vsb + (size_t)(kt + key) * DDIM + d0;
            if (kt + 64 < NTOK) {
                __builtin_prefetch(s1 + 64 * DDIM, 0, 0);
                __builtin_prefetch(s2 + 64 * DDIM, 0, 0);
            }
#pragma unroll
            for (int i = 0; i < 16; ++i) vcT[d0 + i][key] = s1[i];
#pragma unroll
            for (int i = 0; i < 16; ++i) vsT[d0 + i][key] = s2[i];
        }
        wait_async0();   // this wave's async copies landed in LDS
        __syncthreads(); // -> all waves' copies visible

        // S = qc.kc^T + qs.ks^T   (4 key subtiles of 16), K frags from LDS
        v8f S[4];
#pragma unroll
        for (int tt = 0; tt < 4; ++tt) {
            v8f a = {};
            const _Float16* kr = &kcS[tt * 16 + lh][0];
            const _Float16* sr = &ksS[tt * 16 + lh][0];
#pragma unroll
            for (int kk = 0; kk < 2; ++kk) {
                const int off = kk * 32 + 8 * g;
                v16h bc = cat16(*(const v8h*)(kr + off), *(const v8h*)(kr + off + 16));
                a = wmma16(qcf[kk], bc, a);
                v16h bs = cat16(*(const v8h*)(sr + off), *(const v8h*)(sr + off + 16));
                a = wmma16(qsf[kk], bs, a);
            }
            S[tt] = a;
        }

        // online softmax (rows live in 16-lane halves -> xor masks 1..8)
#pragma unroll
        for (int r = 0; r < 8; ++r) {
            float mx = fmaxf(fmaxf(S[0][r], S[1][r]), fmaxf(S[2][r], S[3][r]));
#pragma unroll
            for (int msk = 8; msk >= 1; msk >>= 1)
                mx = fmaxf(mx, __shfl_xor(mx, msk, 32));
            const float mnew  = fmaxf(mrow[r], mx);
            const float alpha = __expf(mrow[r] - mnew);
            float rs = 0.0f;
#pragma unroll
            for (int tt = 0; tt < 4; ++tt) {
                const float p = __expf(S[tt][r] - mnew);
                S[tt][r] = p;
                rs += p;
            }
#pragma unroll
            for (int msk = 8; msk >= 1; msk >>= 1)
                rs += __shfl_xor(rs, msk, 32);
            lrow[r] = lrow[r] * alpha + rs;
            mrow[r] = mnew;
#pragma unroll
            for (int j = 0; j < 4; ++j) { accc[j][r] *= alpha; accs[j][r] *= alpha; }
        }

        // write P (C layout -> row-major LDS strip), wave-private region
#pragma unroll
        for (int tt = 0; tt < 4; ++tt)
#pragma unroll
            for (int r = 0; r < 8; ++r)
                Pld[w][r + 8 * g][tt * 16 + lh] = (_Float16)S[tt][r];

        // DS ops from one wave are in-order; fence the store->load RAW
        asm volatile("s_wait_dscnt 0" ::: "memory");

        // P A-fragments (keys 0-31 and 32-63 of this tile)
        v16h pf[2];
#pragma unroll
        for (int kk = 0; kk < 2; ++kk) {
            const int off = kk * 32 + 8 * g;
            pf[kk] = cat16(*(const v8h*)&Pld[w][lh][off],
                           *(const v8h*)&Pld[w][lh][off + 16]);
        }

        // O += P @ V for both streams (4 d-subtiles)
#pragma unroll
        for (int j = 0; j < 4; ++j) {
            const int drow = j * 16 + lh;
#pragma unroll
            for (int kk = 0; kk < 2; ++kk) {
                const int off = kk * 32 + 8 * g;
                v16h bvc = cat16(*(const v8h*)&vcT[drow][off],
                                 *(const v8h*)&vcT[drow][off + 16]);
                accc[j] = wmma16(pf[kk], bvc, accc[j]);
                v16h bvs = cat16(*(const v8h*)&vsT[drow][off],
                                 *(const v8h*)&vsT[drow][off + 16]);
                accs[j] = wmma16(pf[kk], bvs, accs[j]);
            }
        }
    }

    // epilogue: O /= l, store merged head layout [b*N + n][h*64 + d] (f16)
#pragma unroll
    for (int j = 0; j < 4; ++j) {
        const int d = j * 16 + lh;
#pragma unroll
        for (int r = 0; r < 8; ++r) {
            const int row = q0 + 8 * g + r;
            const float inv = 1.0f / lrow[r];
            const size_t o = ((size_t)(b * NTOK + row)) * CDIM + h * DDIM + d;
            oc[o]  = (_Float16)(accc[j][r] * inv);
            osb[o] = (_Float16)(accs[j][r] * inv);
        }
    }
}

// ---------------------------------------------------------------------------
// Kernel 4: output projection  Y = O @ W + bias   (f16 A, f32 W/bias, f32 out)
// Same tiling as kernel 1; A tile staged via async global->LDS B128.
// ---------------------------------------------------------------------------
__global__ void __launch_bounds__(256)
out_proj_kernel(const _Float16* __restrict__ X,  // [B*N, C] f16
                const float* __restrict__ W,     // [C, C]
                const float* __restrict__ bias,  // [C]
                float* __restrict__ Y)           // [B*N, C]
{
    __shared__ __align__(16) _Float16 As[64][40];
    __shared__ __align__(16) _Float16 Bt[128][40];

    const int t    = threadIdx.x;
    const int lane = t & 31;
    const int w    = t >> 5;
    const int wm   = w & 1;
    const int wn   = w >> 1;
    const int g    = lane >> 4;
    const int lh   = lane & 15;
    const int Mb   = blockIdx.x * 64;
    const int Nb   = blockIdx.y * 128;

    v8f acc[2][2] = {};

    for (int k0 = 0; k0 < CDIM; k0 += 32) {
        {
            const int r = t >> 2, c = (t & 3) * 8;
            // pure f16 copy -> async global->LDS (16 B per thread)
            async_copy_b128(lds_off(&As[r][c]),
                            (uint64_t)(X + (size_t)(Mb + r) * CDIM + k0 + c));
        }
        {
            const int kk = t >> 3, n0 = (t & 7) * 16;
            const float* src = W + (size_t)(k0 + kk) * CDIM + Nb + n0;
            if (k0 + 32 < CDIM) __builtin_prefetch(src + 32 * CDIM, 0, 0);
#pragma unroll
            for (int i = 0; i < 16; ++i) Bt[n0 + i][kk] = (_Float16)src[i];
        }
        wait_async0();
        __syncthreads();

        v16h af[2], bf[2];
#pragma unroll
        for (int ms = 0; ms < 2; ++ms) {
            const int m = wm * 32 + ms * 16 + lh;
            af[ms] = cat16(*(const v8h*)&As[m][8 * g],
                           *(const v8h*)&As[m][8 * g + 16]);
        }
#pragma unroll
        for (int ns = 0; ns < 2; ++ns) {
            const int n = wn * 32 + ns * 16 + lh;
            bf[ns] = cat16(*(const v8h*)&Bt[n][8 * g],
                           *(const v8h*)&Bt[n][8 * g + 16]);
        }
#pragma unroll
        for (int ms = 0; ms < 2; ++ms)
#pragma unroll
            for (int ns = 0; ns < 2; ++ns)
                acc[ms][ns] = wmma16(af[ms], bf[ns], acc[ms][ns]);
        __syncthreads();
    }

#pragma unroll
    for (int ms = 0; ms < 2; ++ms)
#pragma unroll
        for (int ns = 0; ns < 2; ++ns) {
            const int col = Nb + wn * 32 + ns * 16 + lh;
            const float bv = bias[col];
#pragma unroll
            for (int r = 0; r < 8; ++r) {
                const int row = Mb + wm * 32 + ms * 16 + 8 * g + r;
                Y[(size_t)row * CDIM + col] = acc[ms][ns][r] + bv;
            }
        }
}

// ---------------------------------------------------------------------------
// Host launch
// ---------------------------------------------------------------------------
extern "C" void kernel_launch(void* const* d_in, const int* in_sizes, int n_in,
                              void* d_out, int out_size, void* d_ws, size_t ws_size,
                              hipStream_t stream) {
    const float* fc    = (const float*)d_in[0];
    const float* fs    = (const float*)d_in[1];
    const float* Wqc   = (const float*)d_in[2];
    const float* Wqs   = (const float*)d_in[3];
    const float* Wkc   = (const float*)d_in[4];
    const float* Wks   = (const float*)d_in[5];
    const float* Wvc   = (const float*)d_in[6];
    const float* Wvs   = (const float*)d_in[7];
    const float* gamma = (const float*)d_in[8];
    const float* Wpc   = (const float*)d_in[9];
    const float* bpc   = (const float*)d_in[10];
    const float* Wps   = (const float*)d_in[11];
    const float* bps   = (const float*)d_in[12];

    _Float16* ws = (_Float16*)d_ws;
    const size_t SZ = (size_t)BATCH * HEADS * NTOK * DDIM; // == B*N*C elems
    _Float16* qcb = ws;
    _Float16* qsb = ws + 1 * SZ;
    _Float16* kcb = ws + 2 * SZ;
    _Float16* ksb = ws + 3 * SZ;
    _Float16* vcb = ws + 4 * SZ;
    _Float16* vsb = ws + 5 * SZ;
    _Float16* ocb = ws + 6 * SZ;
    _Float16* osb = ws + 7 * SZ;

    const dim3 gemm_grid(BATCH * NTOK / 64, CDIM / 128); // (128, 4)
    proj_qkv_kernel<<<gemm_grid, 256, 0, stream>>>(fc, Wqc, qcb);
    proj_qkv_kernel<<<gemm_grid, 256, 0, stream>>>(fs, Wqs, qsb);
    proj_qkv_kernel<<<gemm_grid, 256, 0, stream>>>(fc, Wkc, kcb);
    proj_qkv_kernel<<<gemm_grid, 256, 0, stream>>>(fs, Wks, ksb);
    proj_qkv_kernel<<<gemm_grid, 256, 0, stream>>>(fc, Wvc, vcb);
    proj_qkv_kernel<<<gemm_grid, 256, 0, stream>>>(fs, Wvs, vsb);

    const int norm_blocks = BATCH * HEADS * NTOK / 8; // one wave per row
    scale_norm_kernel<<<norm_blocks, 256, 0, stream>>>(qcb, gamma, SCALE_QK, 0);
    scale_norm_kernel<<<norm_blocks, 256, 0, stream>>>(qsb, gamma, 1.0f, 1);

    attn_kernel<<<dim3(NTOK / 128, BATCH * HEADS), 256, 0, stream>>>(
        qcb, qsb, kcb, ksb, vcb, vsb, ocb, osb);

    float* out_c = (float*)d_out;
    float* out_s = out_c + SZ;
    out_proj_kernel<<<gemm_grid, 256, 0, stream>>>(ocb, Wpc, bpc, out_c);
    out_proj_kernel<<<gemm_grid, 256, 0, stream>>>(osb, Wps, bps, out_s);
}
